// SeqPPOPolicyRoPEFactorized_28716151341272
// MI455X (gfx1250) — matrix operand order repaired and checked by reading, not compiled, not run on hardware
//
#include <hip/hip_runtime.h>
#include <hip/hip_bf16.h>

// ---------------------------------------------------------------------------
// SeqPPOPolicyRoPEFactorized forward for MI455X (gfx1250, wave32, WMMA).
// bf16 WMMA GEMMs with padded pre-converted weights; fused conv stack in LDS
// with TDM (tensor_load_to_lds) staging of the input image.
//
// d_in layout (setup_inputs() dict order, params flattened depth-first in
// insertion order):
//   0 x_img  1 x_meta  2 legal_action_mask(u8)  3 actions_flat(i32)
//   4..5 c1_w,c1_b  6..7 c2_w,c2_b  8..9 c3_w,c3_b  10..11 fc_w,fc_b
//   12..13 m1_w,m1_b 14..15 m2_w,m2_b 16..17 in_w,in_b 18..19 head_w,head_b
//   20..21 psrc_w,psrc_b 22 src_emb 23..24 dm_w,dm_b 25..26 pdir_w,pdir_b
//   27..28 pmode_w,pmode_b 29..30 v_w,v_b
//   31+12*l: blocks[l] = ln1_g,ln1_b,qkv_w,qkv_b,out_w,out_b,ln2_g,ln2_b,
//                        ff1_w,ff1_b,ff2_w,ff2_b
// d_out: [logp(32) | ent(32) | value(32)] fp32.
// ---------------------------------------------------------------------------

typedef __attribute__((ext_vector_type(16))) __bf16    v16bf;
typedef __attribute__((ext_vector_type(8)))  float     v8f;
typedef __attribute__((ext_vector_type(4)))  unsigned  v4u;
typedef __attribute__((ext_vector_type(8)))  int       v8i;
typedef __attribute__((ext_vector_type(4)))  int       v4i;

static __device__ __forceinline__ __bf16 f2bf(float f) {
  unsigned u = __builtin_bit_cast(unsigned, f);
  unsigned r = u + 0x7FFFu + ((u >> 16) & 1u);   // RNE
  return __builtin_bit_cast(__bf16, (unsigned short)(r >> 16));
}
static __device__ __forceinline__ float bf2f(__bf16 b) {
  unsigned u = ((unsigned)__builtin_bit_cast(unsigned short, b)) << 16;
  return __builtin_bit_cast(float, u);
}

// two 16-byte loads -> one 16-element bf16 fragment
static __device__ __forceinline__ v16bf load_frag16(const __bf16* p0, const __bf16* p1) {
  struct P { v4u a, b; } t;
  t.a = *(const v4u*)p0;
  t.b = *(const v4u*)p1;
  return __builtin_bit_cast(v16bf, t);
}

// ===========================================================================
// fp32 [M][K] -> padded bf16 [Mp][Kp], zero fill.
// ===========================================================================
__global__ void wcvt_kernel(const float* __restrict__ W, __bf16* __restrict__ out,
                            int M, int K, int Mp, int Kp)
{
  int idx = blockIdx.x * blockDim.x + threadIdx.x;
  if (idx >= Mp * Kp) return;
  int m = idx / Kp, k = idx % Kp;
  float v = (m < M && k < K) ? W[(size_t)m * K + k] : 0.f;
  out[idx] = f2bf(v);
}

// conv weights (64, CIN, 3, 3) -> tap-major bf16 [64][9*CINP], kk = tap*CINP+cin
__global__ void wcvt_conv_kernel(const float* __restrict__ W, __bf16* __restrict__ out,
                                 int CIN, int CINP)
{
  int idx = blockIdx.x * blockDim.x + threadIdx.x;
  int Kp = 9 * CINP;
  if (idx >= 64 * Kp) return;
  int m = idx / Kp, kk = idx % Kp;
  int tap = kk / CINP, cin = kk % CINP;
  float v = (cin < CIN) ? W[(size_t)m * CIN * 9 + cin * 9 + tap] : 0.f;
  out[idx] = f2bf(v);
}

// gather z[:, -1] (stride 100*320) -> bf16 [32][320]
__global__ void zlast_kernel(const float* __restrict__ z, __bf16* __restrict__ out)
{
  int idx = blockIdx.x * blockDim.x + threadIdx.x;   // 32*320
  if (idx >= 32 * 320) return;
  int b = idx / 320, j = idx % 320;
  out[idx] = f2bf(z[(size_t)(b * 100 + 99) * 320 + j]);
}

// ===========================================================================
// WMMA GEMM: Y[n,m] = act( X[n,:]·W[m,:] + bias[m] + res[n,m] )
// X: bf16 [N][ldx] (ldx mult of 32). W: bf16 [Mp64][Kp] padded.
// One wave -> 64(M) x 16(N); 8 waves/block. N % 16 == 0.
// ===========================================================================
__global__ void __launch_bounds__(256)
gemm_bf16(const __bf16* __restrict__ X, int ldx,
          const __bf16* __restrict__ W,
          const float* __restrict__ bias,
          const float* __restrict__ res, int ldr,
          float* __restrict__ Yf, __bf16* __restrict__ Yh, int ldy,
          int N, int M, int Kp, int act)
{
  const int lane = threadIdx.x & 31;
  const int wave = threadIdx.x >> 5;
  const int l    = lane & 15;
  const int hi   = lane >> 4;

  const int nbase = blockIdx.x * 16;
  const int mbase = (blockIdx.y * 8 + wave) * 64;
  const int mp64  = (M + 63) & ~63;
  if (mbase >= mp64) return;

  const __bf16* xrow = X + (size_t)(nbase + l) * ldx + hi * 8;
  const __bf16* wr0 = W + (size_t)(mbase +  0 + l) * Kp + hi * 8;
  const __bf16* wr1 = W + (size_t)(mbase + 16 + l) * Kp + hi * 8;
  const __bf16* wr2 = W + (size_t)(mbase + 32 + l) * Kp + hi * 8;
  const __bf16* wr3 = W + (size_t)(mbase + 48 + l) * Kp + hi * 8;

  v8f acc0 = {}, acc1 = {}, acc2 = {}, acc3 = {};
  for (int kb = 0; kb < Kp; kb += 32) {
    v16bf bfrag = load_frag16(xrow + kb, xrow + kb + 16);
    v16bf a0 = load_frag16(wr0 + kb, wr0 + kb + 16);
    v16bf a1 = load_frag16(wr1 + kb, wr1 + kb + 16);
    v16bf a2 = load_frag16(wr2 + kb, wr2 + kb + 16);
    v16bf a3 = load_frag16(wr3 + kb, wr3 + kb + 16);
    acc0 = __builtin_amdgcn_wmma_f32_16x16x32_bf16(false, a0, false, bfrag, (short)0, acc0, false, false);
    acc1 = __builtin_amdgcn_wmma_f32_16x16x32_bf16(false, a1, false, bfrag, (short)0, acc1, false, false);
    acc2 = __builtin_amdgcn_wmma_f32_16x16x32_bf16(false, a2, false, bfrag, (short)0, acc2, false, false);
    acc3 = __builtin_amdgcn_wmma_f32_16x16x32_bf16(false, a3, false, bfrag, (short)0, acc3, false, false);
  }

  const int n = nbase + l;
  if (n >= N) return;
  v8f accs[4] = {acc0, acc1, acc2, acc3};
#pragma unroll
  for (int t = 0; t < 4; ++t) {
#pragma unroll
    for (int r = 0; r < 8; ++r) {
      int m = mbase + t * 16 + r + hi * 8;
      if (m < M) {
        float v = accs[t][r];
        if (bias) v += bias[m];
        if (res)  v += res[(size_t)n * ldr + m];
        if (act)  v = fmaxf(v, 0.f);
        if (Yf) Yf[(size_t)n * ldy + m] = v;
        else    Yh[(size_t)n * ldy + m] = f2bf(v);
      }
    }
  }
}

// ===========================================================================
// Conv step: implicit GEMM, tap-major weights (kk = tap*CINP + cin, CINP pow2).
// Within a 16-element fragment the tap index is constant (span never crosses a
// CINP block), so the im2col gather is base + jj*625 (jj = j + (j&8)) LDS reads + one select.
// One wave handles one pixel-tile x all 64 output channels (4 WMMAs/chunk).
// ===========================================================================
template<int CINP, int CIN>
static __device__ void conv3x3_wmma_lds(const __bf16* __restrict__ in,
                                        const __bf16* __restrict__ Wb,
                                        const float* __restrict__ bias,
                                        __bf16* __restrict__ out)
{
  constexpr int KP = 9 * CINP;
  constexpr int SH = (CINP == 32) ? 5 : 6;
  const int lane = threadIdx.x & 31;
  const int wave = threadIdx.x >> 5;
  const int l    = lane & 15;
  const int hi   = lane >> 4;
  const __bf16 BZ = __builtin_bit_cast(__bf16, (unsigned short)0);

  for (int ntile = wave; ntile < 40; ntile += 8) {
    const int  p   = ntile * 16 + l;
    const bool pok = p < 625;
    const int  py  = p / 25;
    const int  px  = p % 25;

    int  toff[9];
    bool tok[9];
#pragma unroll
    for (int tap = 0; tap < 9; ++tap) {
      int iy = py + tap / 3 - 1;
      int ix = px + tap % 3 - 1;
      bool ok = pok && iy >= 0 && iy < 25 && ix >= 0 && ix < 25;
      tok[tap]  = ok;
      toff[tap] = ok ? (iy * 25 + ix) : 0;
    }

    const __bf16* wr0 = Wb + (size_t)( 0 + l) * KP + hi * 8;
    const __bf16* wr1 = Wb + (size_t)(16 + l) * KP + hi * 8;
    const __bf16* wr2 = Wb + (size_t)(32 + l) * KP + hi * 8;
    const __bf16* wr3 = Wb + (size_t)(48 + l) * KP + hi * 8;

    v8f acc0 = {}, acc1 = {}, acc2 = {}, acc3 = {};
#pragma unroll
    for (int kb = 0; kb < KP; kb += 32) {
      const int kk0  = kb + hi * 8;
      const int tap  = kk0 >> SH;            // constant within fragment
      const int cin0 = kk0 & (CINP - 1);
      const int base = cin0 * 625 + toff[tap];
      const bool okt = tok[tap];
      v16bf b;
#pragma unroll
      for (int j = 0; j < 16; ++j) {
        const int jj = j + (j & 8);
        bool ok = okt && (CINP == CIN || (cin0 + jj) < CIN);
        __bf16 v = in[base + jj * 625];
        b[j] = ok ? v : BZ;
      }
      v16bf a0 = load_frag16(wr0 + kb, wr0 + kb + 16);
      v16bf a1 = load_frag16(wr1 + kb, wr1 + kb + 16);
      v16bf a2 = load_frag16(wr2 + kb, wr2 + kb + 16);
      v16bf a3 = load_frag16(wr3 + kb, wr3 + kb + 16);
      acc0 = __builtin_amdgcn_wmma_f32_16x16x32_bf16(false, a0, false, b, (short)0, acc0, false, false);
      acc1 = __builtin_amdgcn_wmma_f32_16x16x32_bf16(false, a1, false, b, (short)0, acc1, false, false);
      acc2 = __builtin_amdgcn_wmma_f32_16x16x32_bf16(false, a2, false, b, (short)0, acc2, false, false);
      acc3 = __builtin_amdgcn_wmma_f32_16x16x32_bf16(false, a3, false, b, (short)0, acc3, false, false);
    }

    v8f accs[4] = {acc0, acc1, acc2, acc3};
    const int n = ntile * 16 + l;
#pragma unroll
    for (int t = 0; t < 4; ++t) {
#pragma unroll
      for (int r = 0; r < 8; ++r) {
        int m = t * 16 + r + hi * 8;
        if (n < 625) out[m * 625 + n] = f2bf(fmaxf(accs[t][r] + bias[m], 0.f));
      }
    }
  }
}

// ===========================================================================
// Fused conv stack + adaptive pool. One workgroup per image (3200 blocks).
// TDM stages the fp32 image into LDS; activations ping-pong in LDS (bf16).
// Dynamic LDS: 25088(img bf16) + 80000 + 80000 + 50000(fp32 stage) = 235,088 B
// ===========================================================================
#define CONV_LDS_BYTES 235088
#define STAGE_OFF_BYTES 185088

__global__ void __launch_bounds__(256)
conv_stack_kernel(const float* __restrict__ x_img,
                  const __bf16* __restrict__ c1w, const float* __restrict__ c1b,
                  const __bf16* __restrict__ c2w, const float* __restrict__ c2b,
                  const __bf16* __restrict__ c3w, const float* __restrict__ c3b,
                  __bf16* __restrict__ hpool)
{
  extern __shared__ __bf16 smem[];
  __bf16* simg = smem;                        // 20*625 bf16 (pad 12544)
  __bf16* buf1 = smem + 12544;                // 64*625
  __bf16* buf2 = buf1 + 40000;
  float*  stage = (float*)(smem + STAGE_OFF_BYTES / 2);   // 12500 fp32

  const int img = blockIdx.x;
  const float* src = x_img + (size_t)img * 20 * 625;
  const int wave = threadIdx.x >> 5;

  // --- TDM: DMA 50,000 B image into LDS (wave 0 issues, EXEC-independent) ---
  if (wave == 0) {
    unsigned long long ga = (unsigned long long)(const void*)src;
    v4u g0;
    g0[0] = 1u;                                        // count=1, user mode
    g0[1] = (unsigned)STAGE_OFF_BYTES;                 // lds_addr (bytes)
    g0[2] = (unsigned)(ga & 0xFFFFFFFFu);              // global_addr[31:0]
    g0[3] = (unsigned)((ga >> 32) & 0x01FFFFFFu) | (2u << 30);  // [56:32]+type=2
    v8i g1;
    g1[0] = 0x00020000;                                // data_size=2 (4B)
    g1[1] = (int)((12500u & 0xFFFFu) << 16);           // tensor_dim0 lo16
    g1[2] = (int)(1u << 16);                           // dim0 hi=0, tensor_dim1=1 lo
    g1[3] = (int)(12500u << 16);                       // dim1 hi=0, tile_dim0=12500
    g1[4] = 1;                                         // tile_dim1=1, tile_dim2=0
    g1[5] = 12500;                                     // tensor_dim0_stride lo32
    g1[6] = 0;
    g1[7] = 0;
    v4i gz4 = {0, 0, 0, 0};
    v8i gz8 = {0, 0, 0, 0, 0, 0, 0, 0};
    __builtin_amdgcn_tensor_load_to_lds(g0, g1, gz4, gz4, gz8, 0);
    __builtin_amdgcn_s_wait_tensorcnt(0);
  }
  __syncthreads();

  for (int i = threadIdx.x; i < 20 * 625; i += 256) simg[i] = f2bf(stage[i]);
  __syncthreads();

  conv3x3_wmma_lds<32, 20>(simg, c1w, c1b, buf1);
  __syncthreads();
  conv3x3_wmma_lds<64, 64>(buf1, c2w, c2b, buf2);
  __syncthreads();
  conv3x3_wmma_lds<64, 64>(buf2, c3w, c3b, buf1);
  __syncthreads();

  // adaptive 4x4 pool: bins [0,7) [6,13) [12,19) [18,25)
  const int ss[4] = {0, 6, 12, 18};
  const int ee[4] = {7, 13, 19, 25};
  for (int o = threadIdx.x; o < 1024; o += 256) {
    int ch = o >> 4, oy = (o >> 2) & 3, ox = o & 3;
    float s = 0.f;
    for (int y = ss[oy]; y < ee[oy]; ++y)
      for (int x = ss[ox]; x < ee[ox]; ++x)
        s += bf2f(buf1[ch * 625 + y * 25 + x]);
    float cnt = (float)((ee[oy] - ss[oy]) * (ee[ox] - ss[ox]));
    hpool[(size_t)img * 1024 + o] = f2bf(s / cnt);
  }
}

// ===========================================================================
// LayerNorm (fp32 in, bf16 out), one block per row, D=320.
// ===========================================================================
__global__ void __launch_bounds__(128)
layernorm_kernel(const float* __restrict__ X,
                 const float* __restrict__ g, const float* __restrict__ b,
                 __bf16* __restrict__ Y, int D)
{
  __shared__ float red[128];
  const int row = blockIdx.x;
  const float* x = X + (size_t)row * D;

  float s = 0.f;
  for (int i = threadIdx.x; i < D; i += 128) s += x[i];
  red[threadIdx.x] = s; __syncthreads();
  for (int st = 64; st > 0; st >>= 1) {
    if (threadIdx.x < st) red[threadIdx.x] += red[threadIdx.x + st];
    __syncthreads();
  }
  float mu = red[0] / D; __syncthreads();

  float v = 0.f;
  for (int i = threadIdx.x; i < D; i += 128) { float d = x[i] - mu; v += d * d; }
  red[threadIdx.x] = v; __syncthreads();
  for (int st = 64; st > 0; st >>= 1) {
    if (threadIdx.x < st) red[threadIdx.x] += red[threadIdx.x + st];
    __syncthreads();
  }
  float inv = rsqrtf(red[0] / D + 1e-5f);
  for (int i = threadIdx.x; i < D; i += 128)
    Y[(size_t)row * D + i] = f2bf((x[i] - mu) * inv * g[i] + b[i]);
}

// ===========================================================================
// RoPE on q,k inside qkv [3200,960] fp32. hd=40, half=20.
// ===========================================================================
__global__ void rope_kernel(float* __restrict__ qkv)
{
  int idx = blockIdx.x * blockDim.x + threadIdx.x;   // 512000
  if (idx >= 3200 * 8 * 20) return;
  int i  = idx % 20;
  int h  = (idx / 20) & 7;
  int bt = idx / 160;
  int t  = bt % 100;
  float inv = powf(10000.f, -(float)i / 20.f);
  float fr  = (float)t * inv;
  float c = cosf(fr), s = sinf(fr);
#pragma unroll
  for (int off = 0; off <= 320; off += 320) {        // q then k
    size_t base = (size_t)bt * 960 + off + h * 40;
    float z1 = qkv[base + i];
    float z2 = qkv[base + 20 + i];
    qkv[base + i]      = z1 * c - z2 * s;
    qkv[base + 20 + i] = z1 * s + z2 * c;
  }
}

// ===========================================================================
// Attention: one block per (b,h); thread q owns a query row. Out: bf16.
// ===========================================================================
__global__ void __launch_bounds__(128)
attn_kernel(const float* __restrict__ qkv, __bf16* __restrict__ atty)
{
  __shared__ float sc[100 * 100];
  const int b = blockIdx.x >> 3;
  const int h = blockIdx.x & 7;
  const int q = threadIdx.x;
  const float scale = 0.15811388300841898f;          // 40^-0.5
  if (q < 100) {
    const float* qp = qkv + ((size_t)(b * 100 + q)) * 960 + h * 40;
    float mx = -1e30f;
    for (int k = 0; k < 100; ++k) {
      const float* kp = qkv + ((size_t)(b * 100 + k)) * 960 + 320 + h * 40;
      float d = 0.f;
      for (int j = 0; j < 40; ++j) d += qp[j] * kp[j];
      d *= scale;
      sc[q * 100 + k] = d;
      mx = fmaxf(mx, d);
    }
    float S = 0.f;
    for (int k = 0; k < 100; ++k) {
      float e = expf(sc[q * 100 + k] - mx);
      sc[q * 100 + k] = e;
      S += e;
    }
    float invS = 1.f / S;
    __bf16* yo = atty + ((size_t)(b * 100 + q)) * 320 + h * 40;
    for (int j = 0; j < 40; ++j) {
      float a = 0.f;
      for (int k = 0; k < 100; ++k)
        a += sc[q * 100 + k] * qkv[((size_t)(b * 100 + k)) * 960 + 640 + h * 40 + j];
      yo[j] = f2bf(a * invS);
    }
  }
}

// ===========================================================================
// Masked categorical over SRC=625; one block per batch row.
// ===========================================================================
__global__ void __launch_bounds__(256)
src_policy_kernel(const float* __restrict__ logits,
                  const unsigned char* __restrict__ mask,
                  const int* __restrict__ actions,
                  float* __restrict__ lp_at, float* __restrict__ ent_out)
{
  __shared__ float red[256];
  __shared__ float wsrc;
  const int b = blockIdx.x;
  const float* lg = logits + (size_t)b * 625;
  const unsigned char* mk = mask + (size_t)b * 5000;
  const int src = actions[b] / 8;

  float mx = -1e30f;
  for (int s = threadIdx.x; s < 625; s += 256) {
    const unsigned char* r8 = mk + s * 8;
    int any = r8[0] | r8[1] | r8[2] | r8[3] | r8[4] | r8[5] | r8[6] | r8[7];
    float w = any ? lg[s] : -1e9f;
    if (s == src) wsrc = w;
    mx = fmaxf(mx, w);
  }
  red[threadIdx.x] = mx; __syncthreads();
  for (int st = 128; st > 0; st >>= 1) {
    if (threadIdx.x < st) red[threadIdx.x] = fmaxf(red[threadIdx.x], red[threadIdx.x + st]);
    __syncthreads();
  }
  mx = red[0]; __syncthreads();

  float S = 0.f, T = 0.f;
  for (int s = threadIdx.x; s < 625; s += 256) {
    const unsigned char* r8 = mk + s * 8;
    int any = r8[0] | r8[1] | r8[2] | r8[3] | r8[4] | r8[5] | r8[6] | r8[7];
    float w = any ? lg[s] : -1e9f;
    float e = expf(w - mx);
    S += e; T += e * (w - mx);
  }
  red[threadIdx.x] = S; __syncthreads();
  for (int st = 128; st > 0; st >>= 1) {
    if (threadIdx.x < st) red[threadIdx.x] += red[threadIdx.x + st];
    __syncthreads();
  }
  S = red[0]; __syncthreads();
  red[threadIdx.x] = T; __syncthreads();
  for (int st = 128; st > 0; st >>= 1) {
    if (threadIdx.x < st) red[threadIdx.x] += red[threadIdx.x + st];
    __syncthreads();
  }
  T = red[0];
  if (threadIdx.x == 0) {
    float lse = logf(S);
    lp_at[b]   = wsrc - mx - lse;
    ent_out[b] = lse - T / S;
  }
}

// ===========================================================================
// concat(g, src_emb[src]) -> bf16 [32][320]
// ===========================================================================
__global__ void build_ctx_cat(const float* __restrict__ g,
                              const float* __restrict__ emb,
                              const int* __restrict__ actions,
                              __bf16* __restrict__ cat)
{
  int idx = blockIdx.x * blockDim.x + threadIdx.x;   // 32*320
  if (idx >= 32 * 320) return;
  int b = idx / 320, j = idx % 320;
  float v = (j < 256) ? g[(size_t)b * 256 + j]
                      : emb[(size_t)(actions[b] / 8) * 64 + (j - 256)];
  cat[idx] = f2bf(v);
}

// ===========================================================================
// Final heads: value, dir/mode masked categoricals, combine.
// ===========================================================================
static __device__ void masked_cat_small(const float* lg, const bool* m, int n,
                                        int pick, float* lp_at, float* ent)
{
  float mx = -1e30f;
  for (int i = 0; i < n; ++i) { float w = m[i] ? lg[i] : -1e9f; mx = fmaxf(mx, w); }
  float S = 0.f, T = 0.f, wp = 0.f;
  for (int i = 0; i < n; ++i) {
    float w = m[i] ? lg[i] : -1e9f;
    float e = expf(w - mx);
    S += e; T += e * (w - mx);
    if (i == pick) wp = w;
  }
  float lse = logf(S);
  *lp_at = wp - mx - lse;
  *ent   = lse - T / S;
}

__global__ void head_final_kernel(const float* __restrict__ g,
                                  const float* __restrict__ ctx,
                                  const float* __restrict__ vw, const float* __restrict__ vb,
                                  const float* __restrict__ pdw, const float* __restrict__ pdb,
                                  const float* __restrict__ pmw, const float* __restrict__ pmb,
                                  const unsigned char* __restrict__ mask,
                                  const int* __restrict__ actions,
                                  const float* __restrict__ lp_src_at,
                                  const float* __restrict__ ent_src,
                                  float* __restrict__ out)
{
  int b = threadIdx.x;
  if (b >= 32) return;
  const float* gb = g   + (size_t)b * 256;
  const float* cb = ctx + (size_t)b * 256;

  float value = vb[0];
  for (int j = 0; j < 256; ++j) value += vw[j] * gb[j];

  int a = actions[b];
  int src = a / 8, dd = (a % 8) / 2, mo = a % 2;

  float ld[4], lm[2];
  for (int i = 0; i < 4; ++i) {
    float s = pdb[i];
    for (int j = 0; j < 256; ++j) s += pdw[i * 256 + j] * cb[j];
    ld[i] = s;
  }
  for (int i = 0; i < 2; ++i) {
    float s = pmb[i];
    for (int j = 0; j < 256; ++j) s += pmw[i * 256 + j] * cb[j];
    lm[i] = s;
  }

  const unsigned char* rm = mask + (size_t)b * 5000 + src * 8; // row (4,2)
  bool dmask[4], mmask[2];
  for (int i = 0; i < 4; ++i) dmask[i] = (rm[i * 2] | rm[i * 2 + 1]) != 0;
  mmask[0] = (rm[0] | rm[2] | rm[4] | rm[6]) != 0;
  mmask[1] = (rm[1] | rm[3] | rm[5] | rm[7]) != 0;

  float lp_d, ent_d, lp_m, ent_m;
  masked_cat_small(ld, dmask, 4, dd, &lp_d, &ent_d);
  masked_cat_small(lm, mmask, 2, mo, &lp_m, &ent_m);

  out[b]      = lp_src_at[b] + lp_d + lp_m;
  out[32 + b] = ent_src[b] + ent_d + ent_m;
  out[64 + b] = value;
}

// ===========================================================================
// Host orchestration
// ===========================================================================
extern "C" void kernel_launch(void* const* d_in, const int* in_sizes, int n_in,
                              void* d_out, int out_size, void* d_ws, size_t ws_size,
                              hipStream_t stream)
{
  (void)in_sizes; (void)n_in; (void)out_size; (void)ws_size;
  auto F = [&](int i) { return (const float*)d_in[i]; };

  const float* x_img  = F(0);
  const float* x_meta = F(1);
  const unsigned char* lmask = (const unsigned char*)d_in[2];
  const int* actions  = (const int*)d_in[3];

  const float *c1w = F(4),  *c1b = F(5);
  const float *c2w = F(6),  *c2b = F(7);
  const float *c3w = F(8),  *c3b = F(9);
  const float *fcw = F(10), *fcb = F(11);
  const float *m1w = F(12), *m1b = F(13);
  const float *m2w = F(14), *m2b = F(15);
  const float *inw = F(16), *inb = F(17);
  const float *hdw = F(18), *hdb = F(19);
  const float *psw = F(20), *psb = F(21);
  const float *semb = F(22);
  const float *dmw = F(23), *dmb = F(24);
  const float *pdw = F(25), *pdb = F(26);
  const float *pmw = F(27), *pmb = F(28);
  const float *vw  = F(29), *vb  = F(30);

  struct Blk { const float *ln1g,*ln1b,*qkvw,*qkvb,*outw,*outb,*ln2g,*ln2b,*ff1w,*ff1b,*ff2w,*ff2b; };
  Blk blk[4];
  for (int l = 0; l < 4; ++l) {
    int p = 31 + 12 * l;
    blk[l] = { F(p+0),F(p+1),F(p+2),F(p+3),F(p+4),F(p+5),
               F(p+6),F(p+7),F(p+8),F(p+9),F(p+10),F(p+11) };
  }

  // ---- workspace carve: fp32 region then bf16 region ----
  float* wsf = (float*)d_ws;
  size_t offf = 0;
  auto allocf = [&](size_t n) { float* p = wsf + offf; offf += (n + 7) & ~(size_t)7; return p; };
  float* z     = allocf(3200 * 320);
  float* qkv   = allocf(3200 * 960);
  float* gbuf  = allocf(32 * 256);
  float* lsrc  = allocf(32 * 625);
  float* ctx   = allocf(32 * 256);
  float* lp_at = allocf(32);
  float* ent_s = allocf(32);

  __bf16* wsh = (__bf16*)(wsf + offf);
  size_t offh = 0;
  auto alloch = [&](size_t n) { __bf16* p = wsh + offh; offh += (n + 15) & ~(size_t)15; return p; };
  __bf16* hpool  = alloch(3200 * 1024);
  __bf16* meta_h = alloch(3200 * 32);
  __bf16* mbuf   = alloch(3200 * 64);
  __bf16* zin    = alloch(3200 * 320);
  __bf16* h1     = alloch(3200 * 320);
  __bf16* atty   = alloch(3200 * 320);
  __bf16* ffb    = alloch(3200 * 1280);
  __bf16* zlast  = alloch(32 * 320);
  __bf16* gbuf_h = alloch(32 * 256);
  __bf16* catb   = alloch(32 * 320);
  // padded bf16 weights (conv: tap-major [64][9*CINP])
  __bf16* W_c1  = alloch(64 * 288);
  __bf16* W_c2  = alloch(64 * 576);
  __bf16* W_c3  = alloch(64 * 576);
  __bf16* W_fc  = alloch(256 * 1024);
  __bf16* W_m1  = alloch(64 * 32);
  __bf16* W_m2  = alloch(64 * 64);
  __bf16* W_in  = alloch(320 * 320);
  __bf16* W_hd  = alloch(256 * 320);
  __bf16* W_ps  = alloch(640 * 256);
  __bf16* W_dm  = alloch(256 * 320);
  __bf16* W_qkv[4]; __bf16* W_out[4]; __bf16* W_ff1[4]; __bf16* W_ff2[4];
  for (int l = 0; l < 4; ++l) {
    W_qkv[l] = alloch(960 * 320);
    W_out[l] = alloch(320 * 320);
    W_ff1[l] = alloch(1280 * 320);
    W_ff2[l] = alloch(320 * 1280);
  }

  auto cvtw = [&](const float* W, __bf16* out, int M, int K, int Mp, int Kp) {
    int tot = Mp * Kp;
    wcvt_kernel<<<(tot + 255) / 256, 256, 0, stream>>>(W, out, M, K, Mp, Kp);
  };
  auto cvtc = [&](const float* W, __bf16* out, int CIN, int CINP) {
    int tot = 64 * 9 * CINP;
    wcvt_conv_kernel<<<(tot + 255) / 256, 256, 0, stream>>>(W, out, CIN, CINP);
  };
  cvtc(c1w, W_c1, 20, 32);
  cvtc(c2w, W_c2, 64, 64);
  cvtc(c3w, W_c3, 64, 64);
  cvtw(fcw, W_fc, 256, 1024, 256, 1024);
  cvtw(m1w, W_m1, 64, 10, 64, 32);
  cvtw(m2w, W_m2, 64, 64, 64, 64);
  cvtw(inw, W_in, 320, 320, 320, 320);
  cvtw(hdw, W_hd, 256, 320, 256, 320);
  cvtw(psw, W_ps, 625, 256, 640, 256);
  cvtw(dmw, W_dm, 256, 320, 256, 320);
  for (int l = 0; l < 4; ++l) {
    cvtw(blk[l].qkvw, W_qkv[l], 960, 320, 960, 320);
    cvtw(blk[l].outw, W_out[l], 320, 320, 320, 320);
    cvtw(blk[l].ff1w, W_ff1[l], 1280, 320, 1280, 320);
    cvtw(blk[l].ff2w, W_ff2[l], 320, 1280, 320, 1280);
  }
  cvtw(x_meta, meta_h, 3200, 10, 3200, 32);

  auto gemm = [&](const __bf16* X, int ldx, const __bf16* Wb, int Kp,
                  const float* bias, const float* res, int ldr,
                  float* Yf, __bf16* Yh, int ldy, int N, int M, int act) {
    int mp64 = (M + 63) & ~63;
    dim3 g(N / 16, (mp64 / 64 + 7) / 8);
    gemm_bf16<<<g, 256, 0, stream>>>(X, ldx, Wb, bias, res, ldr, Yf, Yh, ldy, N, M, Kp, act);
  };

  // --- fused conv stack + pool (TDM staging) ---
  conv_stack_kernel<<<3200, 256, CONV_LDS_BYTES, stream>>>(
      x_img, W_c1, c1b, W_c2, c2b, W_c3, c3b, hpool);

  // --- frame fc + meta MLP -> zin[3200,320] (bf16) ---
  gemm(hpool, 1024, W_fc, 1024, fcb, nullptr, 0, nullptr, zin, 320, 3200, 256, 1);
  gemm(meta_h, 32, W_m1, 32, m1b, nullptr, 0, nullptr, mbuf, 64, 3200, 64, 1);
  gemm(mbuf, 64, W_m2, 64, m2b, nullptr, 0, nullptr, zin + 256, 320, 3200, 64, 1);
  // --- input projection -> z (fp32) ---
  gemm(zin, 320, W_in, 320, inb, nullptr, 0, z, nullptr, 320, 3200, 320, 0);

  // --- transformer blocks ---
  for (int l = 0; l < 4; ++l) {
    layernorm_kernel<<<3200, 128, 0, stream>>>(z, blk[l].ln1g, blk[l].ln1b, h1, 320);
    gemm(h1, 320, W_qkv[l], 320, blk[l].qkvb, nullptr, 0, qkv, nullptr, 960, 3200, 960, 0);
    rope_kernel<<<(512000 + 255) / 256, 256, 0, stream>>>(qkv);
    attn_kernel<<<256, 128, 0, stream>>>(qkv, atty);
    gemm(atty, 320, W_out[l], 320, blk[l].outb, z, 320, z, nullptr, 320, 3200, 320, 0);
    layernorm_kernel<<<3200, 128, 0, stream>>>(z, blk[l].ln2g, blk[l].ln2b, h1, 320);
    gemm(h1, 320, W_ff1[l], 320, blk[l].ff1b, nullptr, 0, nullptr, ffb, 1280, 3200, 1280, 1);
    gemm(ffb, 1280, W_ff2[l], 1280, blk[l].ff2b, z, 320, z, nullptr, 320, 3200, 320, 0);
  }

  // --- heads (B=32) ---
  zlast_kernel<<<(32 * 320 + 255) / 256, 256, 0, stream>>>(z, zlast);
  gemm(zlast, 320, W_hd, 320, hdb, nullptr, 0, gbuf, nullptr, 256, 32, 256, 1);
  wcvt_kernel<<<(32 * 256 + 255) / 256, 256, 0, stream>>>(gbuf, gbuf_h, 32, 256, 32, 256);
  gemm(gbuf_h, 256, W_ps, 256, psb, nullptr, 0, lsrc, nullptr, 625, 32, 625, 0);
  src_policy_kernel<<<32, 256, 0, stream>>>(lsrc, lmask, actions, lp_at, ent_s);
  build_ctx_cat<<<(32 * 320 + 255) / 256, 256, 0, stream>>>(gbuf, semb, actions, catb);
  gemm(catb, 320, W_dm, 320, dmb, nullptr, 0, ctx, nullptr, 256, 32, 256, 1);
  head_final_kernel<<<1, 32, 0, stream>>>(gbuf, ctx, vw, vb, pdw, pdb, pmw, pmb,
                                          lmask, actions, lp_at, ent_s, (float*)d_out);
}
